// HL_HGCNN_68702296866882
// MI455X (gfx1250) — matrix operand order repaired and checked by reading, not compiled, not run on hardware
//
#include <hip/hip_runtime.h>
#include <hip/hip_bf16.h>
#include <stdint.h>

// ---------------------------------------------------------------------------
// HL_HGCNN forward for MI455X (gfx1250, wave32, WMMA).
// GEMMs: bf16 WMMA 16x16x32, f32 accum; swizzled + double-buffered LDS so
// fragments are single contiguous b128 LDS loads; tile staging uses
// GLOBAL_LOAD_ASYNC_TO_LDS_B128 (ASYNCcnt) when the toolchain exposes it,
// else vectorized load/store. Sparse parts: HW fp32 atomics. BN+ReLU+
// residual+bf16-cast fused into one elementwise pass.
// ---------------------------------------------------------------------------

typedef __attribute__((ext_vector_type(16))) __bf16 v16bf;
typedef __attribute__((ext_vector_type(8)))  __bf16 v8bf;
typedef __attribute__((ext_vector_type(8)))  float  v8f;
typedef __attribute__((ext_vector_type(4)))  int    i32x4;

#if defined(__AMDGCN__) && __has_builtin(__builtin_amdgcn_global_load_async_to_lds_b128) && \
    __has_builtin(__builtin_amdgcn_s_wait_asynccnt)
#define HAS_ASYNC_LDS 1
#else
#define HAS_ASYNC_LDS 0
#endif

// memory->LDS 16-byte move: async (no VGPR round-trip, ASYNCcnt) if available
__device__ __forceinline__ void lds_copy16(const __bf16* gp, __bf16* lp) {
#if HAS_ASYNC_LDS
  __builtin_amdgcn_global_load_async_to_lds_b128(
      (__attribute__((address_space(1))) i32x4*)(void*)(const_cast<__bf16*>(gp)),
      (__attribute__((address_space(3))) i32x4*)(void*)lp, 0, 0);
#else
  *reinterpret_cast<uint4*>(lp) = *reinterpret_cast<const uint4*>(gp);
#endif
}

__device__ __forceinline__ void stage_fence() {
#if HAS_ASYNC_LDS
  __builtin_amdgcn_s_wait_asynccnt(0);
#endif
  __syncthreads();
}

// ----------------------------- elementwise ---------------------------------

__global__ void fill_kernel(float* __restrict__ p, float v, long long n) {
  long long i = (long long)blockIdx.x * blockDim.x + threadIdx.x;
  if (i < n) p[i] = v;
}

__global__ void cast_kernel(const float* __restrict__ in, __bf16* __restrict__ out, long long n) {
  long long i = (long long)blockIdx.x * blockDim.x + threadIdx.x;
  if (i < n) out[i] = (__bf16)in[i];
}

// weight cast WITH transpose: in [K][N] f32 -> out [N][K] bf16, so B-tile
// staging reads K-contiguous 16-byte chunks (async-to-LDS friendly).
__global__ void cast_transpose_kernel(const float* __restrict__ in, __bf16* __restrict__ out,
                                      int K, int N) {
  long long i = (long long)blockIdx.x * blockDim.x + threadIdx.x;
  if (i < (long long)K * N) {
    int k = (int)(i / N), n = (int)(i % N);
    out[(long long)n * K + k] = (__bf16)in[i];
  }
}

// f32 [M,F] -> bf16 sub-block of [M,ldo] at column offset colofs
__global__ void cast_copy_cols_kernel(const float* __restrict__ in, __bf16* __restrict__ out,
                                      long long M, int F, int ldo, int colofs) {
  long long i = (long long)blockIdx.x * blockDim.x + threadIdx.x;
  if (i < M * (long long)F) {
    long long r = i / F; int c = (int)(i % F);
    out[r * ldo + colofs + c] = (__bf16)in[i];
  }
}

// ----------------------------- sparse ops ----------------------------------

// out[dst] += -w * x[src]   (out pre-initialized to x => T1 = x - L x)
__global__ void spmm_axpy_kernel(const int* __restrict__ src, const int* __restrict__ dst,
                                 const float* __restrict__ w, const float* __restrict__ x,
                                 float* __restrict__ out, int nnz, int F) {
  long long i = (long long)blockIdx.x * blockDim.x + threadIdx.x;
  if (i < (long long)nnz * F) {
    int e = (int)(i / F), c = (int)(i % F);
    atomicAdd(&out[(long long)dst[e] * F + c], -w[e] * x[(long long)src[e] * F + c]);
  }
}

__global__ void degree_kernel(const int* __restrict__ flat, float* __restrict__ D, int n) {
  int i = blockIdx.x * blockDim.x + threadIdx.x;
  if (i < n) atomicAdd(&D[flat[i]], 1.0f);
}

// scatter edge features to both endpoint nodes
__global__ void e2n_kernel(const float* __restrict__ xs, const int* __restrict__ src,
                           const int* __restrict__ dst, float* __restrict__ out, int S_, int F) {
  long long i = (long long)blockIdx.x * blockDim.x + threadIdx.x;
  if (i < (long long)S_ * F) {
    int e = (int)(i / F), c = (int)(i % F);
    float v = xs[i];
    atomicAdd(&out[(long long)src[e] * F + c], v);
    atomicAdd(&out[(long long)dst[e] * F + c], v);
  }
}

__global__ void divdeg_cast_kernel(const float* __restrict__ tmp, const float* __restrict__ D,
                                   __bf16* __restrict__ out, long long M, int F, int ldo, int colofs) {
  long long i = (long long)blockIdx.x * blockDim.x + threadIdx.x;
  if (i < M * (long long)F) {
    long long r = i / F; int c = (int)(i % F);
    out[r * ldo + colofs + c] = (__bf16)(tmp[i] / D[r]);
  }
}

// gather-average endpoint node features per edge, straight to bf16 concat
__global__ void n2e_cast_kernel(const float* __restrict__ xt, const int* __restrict__ src,
                                const int* __restrict__ dst, __bf16* __restrict__ out,
                                int S_, int F, int ldo, int colofs) {
  long long i = (long long)blockIdx.x * blockDim.x + threadIdx.x;
  if (i < (long long)S_ * F) {
    int e = (int)(i / F), c = (int)(i % F);
    float v = 0.5f * (xt[(long long)src[e] * F + c] + xt[(long long)dst[e] * F + c]);
    out[(long long)e * ldo + colofs + c] = (__bf16)v;
  }
}

// ----------------------------- batchnorm -----------------------------------

// pass 1: coalesced partial sums. block = 64 cols x 4 row-threads,
// grid = (F/64, ROWCHUNKS). Reads 64 consecutive floats per row group.
__global__ void bn_stats_partial_kernel(const float* __restrict__ x, float* __restrict__ sum,
                                        float* __restrict__ sumsq, long long M, int F) {
  int c = blockIdx.x * 64 + (threadIdx.x & 63);
  int rt = threadIdx.x >> 6;  // 0..3
  long long rows_per = (M + gridDim.y - 1) / gridDim.y;
  long long r0 = (long long)blockIdx.y * rows_per;
  long long r1 = r0 + rows_per; if (r1 > M) r1 = M;
  float s = 0.f, s2 = 0.f;
  for (long long r = r0 + rt; r < r1; r += 4) {
    float v = x[r * F + c];
    s += v; s2 += v * v;
  }
  __shared__ float sh[256], sh2[256];
  sh[threadIdx.x] = s; sh2[threadIdx.x] = s2; __syncthreads();
  if (rt == 0) {
    int t = threadIdx.x;
    s  = sh[t]  + sh[t + 64]  + sh[t + 128]  + sh[t + 192];
    s2 = sh2[t] + sh2[t + 64] + sh2[t + 128] + sh2[t + 192];
    atomicAdd(&sum[c], s);
    atomicAdd(&sumsq[c], s2);
  }
}

__global__ void bn_finalize_kernel(const float* __restrict__ sum, const float* __restrict__ sumsq,
                                   float* __restrict__ mean, float* __restrict__ var,
                                   long long M, int F) {
  int c = blockIdx.x * blockDim.x + threadIdx.x;
  if (c < F) {
    float mu = sum[c] / M;
    mean[c] = mu;
    var[c] = sumsq[c] / M - mu * mu;
  }
}

// fused BN + ReLU + alpha-residual + f32/bf16 dual write
__global__ void bn_relu_kernel(const float* __restrict__ y, const float* __restrict__ mean,
                               const float* __restrict__ var, const float* __restrict__ gamma,
                               const float* __restrict__ beta, const float* __restrict__ x0,
                               float alpha, float* __restrict__ xo, __bf16* __restrict__ xob,
                               long long M, int F) {
  long long i = (long long)blockIdx.x * blockDim.x + threadIdx.x;
  if (i < M * (long long)F) {
    int c = (int)(i % F);
    float v = (y[i] - mean[c]) * rsqrtf(var[c] + 1e-5f) * gamma[c] + beta[c];
    v = fmaxf(v, 0.f);
    if (x0) v = (1.f - alpha) * v + alpha * x0[i];
    xo[i] = v;
    xob[i] = (__bf16)v;
  }
}

// ----------------------------- WMMA GEMM -----------------------------------
// LDS tiles are stored pre-swizzled in WMMA fragment order:
//   As[buf][(wave*32 + lane)*16 + e]   e=0..7  -> K = kbase + half*8 + e
//                                      e=8..15 -> K = kbase + 16 + half*8 + (e-8)
//   Bs[buf][(sub *32 + lane)*16 + e]   same K order as As
// so every fragment is ONE contiguous 32-byte LDS read (2 x ds_load_b128),
// and every staged group is ONE contiguous 16-byte mem->LDS move (async).
// Wt is [N][K] bf16 (pre-transposed by cast_transpose_kernel).
// C[M,N] (+)= A[M,K] @ W[K,N] + bias. Block 256 thr = 8 waves, tile 128x64,
// BK=32, double-buffered (one barrier per K-step).
__global__ void __launch_bounds__(256)
hgcnn_wmma_gemm_bf16(const __bf16* __restrict__ A, const __bf16* __restrict__ Wt,
                     const float* __restrict__ bias, float* __restrict__ C,
                     int M, int K, int N, int accum) {
  __shared__ __bf16 As[2][8 * 32 * 16];   // 8 KB per buffer
  __shared__ __bf16 Bs[2][4 * 32 * 16];   // 4 KB per buffer

  const int tid  = threadIdx.x;
  const int wave = tid >> 5;
  const int lane = tid & 31;
  const int m0 = blockIdx.x * 128;
  const int n0 = blockIdx.y * 64;
  const bool fastK = ((K & 7) == 0);
  const bool fastA = fastK && (m0 + 128 <= M);

  auto stageA = [&](int k0, int buf) {
    if (fastA && (k0 + 32 <= K)) {
      // 512 groups of 8 cols; one 16-byte mem->LDS move per group
#pragma unroll
      for (int g = tid; g < 512; g += 256) {
        int row = g >> 2, cg = (g & 3) * 8;
        int w = row >> 4, r = row & 15;
        int half = (cg >> 3) & 1, chunk = cg >> 4;
        lds_copy16(&A[(long long)(m0 + row) * K + k0 + cg],
                   &As[buf][(w * 32 + half * 16 + r) * 16 + chunk * 8]);
      }
    } else {
      for (int i = tid; i < 128 * 32; i += 256) {
        int row = i >> 5, c = i & 31;
        int gr = m0 + row, gc = k0 + c;
        __bf16 v = (gr < M && gc < K) ? A[(long long)gr * K + gc] : (__bf16)0.0f;
        int w = row >> 4, r = row & 15;
        int half = (c >> 3) & 1, chunk = c >> 4;
        As[buf][(w * 32 + half * 16 + r) * 16 + chunk * 8 + (c & 7)] = v;
      }
    }
  };

  auto stageB = [&](int k0, int buf) {
    if (fastK && (k0 + 32 <= K)) {
      // 256 groups (1/thread): Wt row n, 8 K-contiguous -> 16-byte LDS chunk
      int n = tid >> 2, kg = (tid & 3) * 8;
      lds_copy16(&Wt[(long long)(n0 + n) * K + k0 + kg],
                 &Bs[buf][((n >> 4) * 32 + ((kg >> 3) & 1) * 16 + (n & 15)) * 16 + (kg >> 4) * 8]);
    } else {
      for (int i = tid; i < 32 * 64; i += 256) {
        int k = i >> 6, n = i & 63;
        __bf16 v = (k0 + k < K) ? Wt[(long long)(n0 + n) * K + k0 + k] : (__bf16)0.0f;
        Bs[buf][((n >> 4) * 32 + ((k >> 3) & 1) * 16 + (n & 15)) * 16 + (k >> 4) * 8 + (k & 7)] = v;
      }
    }
  };

  v8f zero = {};
  v8f acc[4] = {zero, zero, zero, zero};

  stageA(0, 0);
  stageB(0, 0);
  stage_fence();

  int buf = 0;
  for (int k0 = 0; k0 < K; k0 += 32) {
    int kn = k0 + 32;
    if (kn < K) {
      stageA(kn, buf ^ 1);                         // streams in during compute
      stageB(kn, buf ^ 1);
      __builtin_prefetch(&A[(long long)(m0 + (tid >> 1)) * K + kn], 0, 1);
    }
    v16bf a = *reinterpret_cast<const v16bf*>(&As[buf][(wave * 32 + lane) * 16]);
#pragma unroll
    for (int j = 0; j < 4; ++j) {
      v16bf b = *reinterpret_cast<const v16bf*>(&Bs[buf][(j * 32 + lane) * 16]);
      acc[j] = __builtin_amdgcn_wmma_f32_16x16x32_bf16(
          false, a, false, b, (short)0, acc[j], false, false);
    }
    stage_fence();
    buf ^= 1;
  }

  // writeback: VGPR r holds M=r (lanes 0-15) / M=r+8 (lanes 16-31), N=lane&15
  const int half = lane >> 4, col = lane & 15;
#pragma unroll
  for (int j = 0; j < 4; ++j) {
    int gc = n0 + j * 16 + col;
#pragma unroll
    for (int r = 0; r < 8; ++r) {
      int gr = m0 + wave * 16 + r + half * 8;
      if (gr < M) {
        float v = acc[j][r] + bias[gc];
        long long o = (long long)gr * N + gc;
        if (accum) v += C[o];
        C[o] = v;
      }
    }
  }
}

// ----------------------------- pooling / head ------------------------------

__global__ void pool_kernel(const float* __restrict__ x, float* __restrict__ out,
                            int rowsPer, int F, int ldo, int colofs) {
  int b = blockIdx.x / F, c = blockIdx.x % F, tid = threadIdx.x;
  const float* xb = x + (long long)b * rowsPer * F;
  float s = 0.f;
  for (int r = tid; r < rowsPer; r += blockDim.x) s += xb[(long long)r * F + c];
  __shared__ float sh[256];
  sh[tid] = s; __syncthreads();
  for (int o = 128; o > 0; o >>= 1) { if (tid < o) sh[tid] += sh[tid + o]; __syncthreads(); }
  if (tid == 0) out[b * ldo + colofs + c] = sh[0] / rowsPer;
}

__global__ void final_linear_kernel(const float* __restrict__ pool, const float* __restrict__ W,
                                    const float* __restrict__ bias, float* __restrict__ out,
                                    int Bn, int Kd, int Nd) {
  int i = blockIdx.x * blockDim.x + threadIdx.x;
  if (i < Bn * Nd) {
    int b = i / Nd, o = i % Nd;
    float s = bias[o];
    for (int k = 0; k < Kd; ++k) s += pool[b * Kd + k] * W[k * Nd + o];
    out[i] = s;
  }
}

// ----------------------------- host orchestration --------------------------

struct Lin { const float* W; const float* b; };
struct CBR { const float* beta; const float* gamma; Lin lin[2]; int K; };

extern "C" void kernel_launch(void* const* d_in, const int* in_sizes, int n_in,
                              void* d_out, int out_size, void* d_ws, size_t ws_size,
                              hipStream_t stream) {
  (void)in_sizes; (void)n_in; (void)out_size; (void)ws_size;

  // ---- model constants ----
  const int Bg = 16, NODE = 1024, EDGE = 3906;
  const int NN = Bg * NODE;        // 16384 nodes
  const int S  = Bg * EDGE;        // 62496 edges
  const int ET = NN * 6, ES = S * 4;
  const int FILT[4] = {64, 128, 256, 512};

  // ---- walk the jax pytree leaves (dicts flatten in sorted-key order) ----
  int idx = 0;
  auto nextf = [&]() { return (const float*)d_in[idx++]; };
  auto read_lin = [&](Lin& L) { L.W = nextf(); L.b = nextf(); };
  auto read_cbr = [&](CBR& c, int K) {
    c.beta = nextf(); c.gamma = nextf(); c.K = K;      // 'bn' < 'conv'
    for (int k = 0; k < K; ++k) read_lin(c.lin[k]);
  };

  CBR EC[4][2], NC[4][2], ECin, NCin, ie2n[3][2], in2e[3][2];
  Lin e0p[3], n0p[3], outlin;
  for (int i = 0; i < 4; ++i) for (int j = 0; j < 2; ++j) read_cbr(EC[i][j], 2);   // EC0..EC3
  read_cbr(ECin, 2);                                                              // EC_in
  for (int i = 0; i < 4; ++i) for (int j = 0; j < 2; ++j) read_cbr(NC[i][j], 2);  // NC0..NC3
  read_cbr(NCin, 2);                                                              // NC_in
  for (int i = 0; i < 3; ++i) read_lin(e0p[i]);                                   // e0_proj*
  for (int i = 0; i < 3; ++i) for (int j = 0; j < 2; ++j) read_cbr(ie2n[i][j], 1);
  for (int i = 0; i < 3; ++i) for (int j = 0; j < 2; ++j) read_cbr(in2e[i][j], 1);
  for (int i = 0; i < 3; ++i) read_lin(n0p[i]);                                   // n0_proj*
  read_lin(outlin);                                                               // out

  const float* x_s_in = nextf();
  const int*   ei_s   = (const int*)d_in[idx++];
  const float* ew_s   = nextf();
  const float* x_t_in = nextf();
  const int*   ei_t   = (const int*)d_in[idx++];
  const float* ew_t   = nextf();
  const int*   einc   = (const int*)d_in[idx++];

  // ---- workspace bump allocator ----
  uintptr_t wp = (uintptr_t)d_ws;
  auto alloc = [&](size_t bytes) {
    wp = (wp + 255) & ~(uintptr_t)255;
    void* p = (void*)wp; wp += bytes; return p;
  };
  float*  xs   = (float*)alloc((size_t)S * 512 * 4);
  __bf16* xsb  = (__bf16*)alloc((size_t)S * 512 * 2);
  float*  xs0  = (float*)alloc((size_t)S * 512 * 4);
  __bf16* xs0b = (__bf16*)alloc((size_t)S * 512 * 2);
  float*  ys   = (float*)alloc((size_t)S * 512 * 4);
  float*  xt   = (float*)alloc((size_t)NN * 512 * 4);
  __bf16* xtb  = (__bf16*)alloc((size_t)NN * 512 * 2);
  float*  xt0  = (float*)alloc((size_t)NN * 512 * 4);
  __bf16* xt0b = (__bf16*)alloc((size_t)NN * 512 * 2);
  float*  yt   = (float*)alloc((size_t)NN * 512 * 4);
  float*  t1   = (float*)alloc((size_t)S * 512 * 4);   // shared T1 scratch
  __bf16* t1b  = (__bf16*)alloc((size_t)S * 512 * 2);
  __bf16* cats = (__bf16*)alloc((size_t)S * 1024 * 2);
  __bf16* catt = (__bf16*)alloc((size_t)NN * 1024 * 2);
  float*  tmpx = (float*)alloc((size_t)NN * 512 * 4);
  float*  Deg  = (float*)alloc((size_t)NN * 4);
  float*  mean = (float*)alloc(512 * 4);
  float*  var  = (float*)alloc(512 * 4);
  float*  sums = (float*)alloc(512 * 4);
  float*  sqs  = (float*)alloc(512 * 4);
  __bf16* wbf  = (__bf16*)alloc((size_t)1024 * 512 * 2);   // transposed weights
  float*  pool = (float*)alloc((size_t)Bg * 1024 * 4);

  auto B1 = [](long long n) { return dim3((unsigned)((n + 255) / 256)); };

  // GEMM: cast+transpose weights to bf16 scratch, then WMMA tile kernel
  auto gemm = [&](const __bf16* A, const float* Wf, const float* bias, float* C,
                  int M, int Kd, int Nd, int accum) {
    cast_transpose_kernel<<<B1((long long)Kd * Nd), 256, 0, stream>>>(Wf, wbf, Kd, Nd);
    dim3 grid((M + 127) / 128, Nd / 64);
    hgcnn_wmma_gemm_bf16<<<grid, 256, 0, stream>>>(A, wbf, bias, C, M, Kd, Nd, accum);
  };

  // T1 = x - L x  (copy + negative scatter-add), plus bf16 shadow
  auto make_t1 = [&](const float* x, const int* ei, const float* ew, int nnz, int M, int F) {
    (void)hipMemcpyAsync(t1, x, (size_t)M * F * 4, hipMemcpyDeviceToDevice, stream);
    spmm_axpy_kernel<<<B1((long long)nnz * F), 256, 0, stream>>>(ei, ei + nnz, ew, x, t1, nnz, F);
    cast_kernel<<<B1((long long)M * F), 256, 0, stream>>>(t1, t1b, (long long)M * F);
  };

  // full conv -> BN -> ReLU (-> alpha blend) block
  auto cbr_apply = [&](const CBR& c, const float* x, const __bf16* xb,
                       const int* ei, const float* ew, int nnz, int M, int din, int dout,
                       const float* x0, float alpha, float* xo, __bf16* xob, float* y) {
    gemm(xb, c.lin[0].W, c.lin[0].b, y, M, din, dout, 0);
    if (c.K > 1) {
      make_t1(x, ei, ew, nnz, M, din);
      gemm(t1b, c.lin[1].W, c.lin[1].b, y, M, din, dout, 1);
    }
    fill_kernel<<<1, 256, 0, stream>>>(sums, 0.f, dout);
    fill_kernel<<<1, 256, 0, stream>>>(sqs, 0.f, dout);
    bn_stats_partial_kernel<<<dim3(dout / 64, 32), 256, 0, stream>>>(y, sums, sqs, M, dout);
    bn_finalize_kernel<<<B1(dout), 256, 0, stream>>>(sums, sqs, mean, var, M, dout);
    bn_relu_kernel<<<B1((long long)M * dout), 256, 0, stream>>>(
        y, mean, var, c.gamma, c.beta, x0, alpha, xo, xob, (long long)M, dout);
  };

  // ---- node degrees (constant across layers) ----
  fill_kernel<<<B1(NN), 256, 0, stream>>>(Deg, 0.f, NN);
  degree_kernel<<<B1(2 * S), 256, 0, stream>>>(einc, Deg, 2 * S);

  // ---- input convs ----
  cast_kernel<<<B1((long long)S * 9), 256, 0, stream>>>(x_s_in, xsb, (long long)S * 9);
  cbr_apply(ECin, x_s_in, xsb, ei_s, ew_s, ES, S, 9, 64, nullptr, 0.f, xs, xsb, ys);
  cast_kernel<<<B1((long long)NN * 5), 256, 0, stream>>>(x_t_in, xtb, (long long)NN * 5);
  cbr_apply(NCin, x_t_in, xtb, ei_t, ew_t, ET, NN, 5, 64, nullptr, 0.f, xt, xtb, yt);

  (void)hipMemcpyAsync(xs0, xs, (size_t)S * 64 * 4, hipMemcpyDeviceToDevice, stream);
  (void)hipMemcpyAsync(xs0b, xsb, (size_t)S * 64 * 2, hipMemcpyDeviceToDevice, stream);
  (void)hipMemcpyAsync(xt0, xt, (size_t)NN * 64 * 4, hipMemcpyDeviceToDevice, stream);
  (void)hipMemcpyAsync(xt0b, xtb, (size_t)NN * 64 * 2, hipMemcpyDeviceToDevice, stream);

  // ---- main stages ----
  int fin = 64;
  for (int i = 0; i < 4; ++i) {
    int f = FILT[i];
    for (int j = 0; j < 2; ++j) {
      int dj = (j == 0) ? fin : f;
      cbr_apply(NC[i][j], xt, xtb, ei_t, ew_t, ET, NN, dj, f, xt0, 0.5f, xt, xtb, yt);
    }
    for (int j = 0; j < 2; ++j) {
      int dj = (j == 0) ? fin : f;
      cbr_apply(EC[i][j], xs, xsb, ei_s, ew_s, ES, S, dj, f, xs0, 0.5f, xs, xsb, ys);
    }

    if (i < 3) {
      int f2 = 2 * f;
      // temp_xt = (scatter xs to endpoints) / D  -> concat_t[:, f:2f]
      fill_kernel<<<B1((long long)NN * f), 256, 0, stream>>>(tmpx, 0.f, (long long)NN * f);
      e2n_kernel<<<B1((long long)S * f), 256, 0, stream>>>(xs, einc, einc + S, tmpx, S, f);
      cast_copy_cols_kernel<<<B1((long long)NN * f), 256, 0, stream>>>(xt, catt, NN, f, f2, 0);
      divdeg_cast_kernel<<<B1((long long)NN * f), 256, 0, stream>>>(tmpx, Deg, catt, NN, f, f2, f);
      // temp_xs = mean of endpoint node features -> concat_s[:, f:2f]
      cast_copy_cols_kernel<<<B1((long long)S * f), 256, 0, stream>>>(xs, cats, S, f, f2, 0);
      n2e_cast_kernel<<<B1((long long)S * f), 256, 0, stream>>>(xt, einc, einc + S, cats, S, f, f2, f);

      // interface convs (K=1: pure GEMM + BN + ReLU)
      cbr_apply(ie2n[i][0], nullptr, catt, ei_t, ew_t, ET, NN, f2, f, nullptr, 0.f, xt, xtb, yt);
      cbr_apply(ie2n[i][1], nullptr, xtb, ei_t, ew_t, ET, NN, f, f, nullptr, 0.f, xt, xtb, yt);
      cbr_apply(in2e[i][0], nullptr, cats, ei_s, ew_s, ES, S, f2, f, nullptr, 0.f, xs, xsb, ys);
      cbr_apply(in2e[i][1], nullptr, xsb, ei_s, ew_s, ES, S, f, f, nullptr, 0.f, xs, xsb, ys);

      // residual projections (K=1 lag_conv, no BN/ReLU)
      int fn = FILT[i + 1];
      gemm(xt0b, n0p[i].W, n0p[i].b, yt, NN, f, fn, 0);
      (void)hipMemcpyAsync(xt0, yt, (size_t)NN * fn * 4, hipMemcpyDeviceToDevice, stream);
      cast_kernel<<<B1((long long)NN * fn), 256, 0, stream>>>(xt0, xt0b, (long long)NN * fn);
      gemm(xs0b, e0p[i].W, e0p[i].b, ys, S, f, fn, 0);
      (void)hipMemcpyAsync(xs0, ys, (size_t)S * fn * 4, hipMemcpyDeviceToDevice, stream);
      cast_kernel<<<B1((long long)S * fn), 256, 0, stream>>>(xs0, xs0b, (long long)S * fn);
    }
    fin = f;
  }

  // ---- global mean pool + classifier head ----
  pool_kernel<<<Bg * 512, 256, 0, stream>>>(xs, pool, EDGE, 512, 1024, 0);
  pool_kernel<<<Bg * 512, 256, 0, stream>>>(xt, pool, NODE, 512, 1024, 512);
  final_linear_kernel<<<1, 256, 0, stream>>>(pool, outlin.W, outlin.b, (float*)d_out,
                                             Bg, 1024, 10);
}
